// QNetwork_75239237091459
// MI455X (gfx1250) — matrix operand order repaired and checked by reading, not compiled, not run on hardware
//
#include <hip/hip_runtime.h>
#include <hip/hip_bf16.h>

// ---------------------------------------------------------------------------
// Spiking QNetwork forward for MI455X (gfx1250, wave32).
// GEMMs: V_WMMA_F32_16X16X32_F16 (spikes {0,1} exact in f16, f32 accum).
// Shared A-operand (spike tiles) staged in LDS via async global->LDS copies
// (double-buffered, ASYNCcnt), consumed via ds_load_b128 fragments.
// Frames serialized by kernel launches (graph-capture friendly).
// ---------------------------------------------------------------------------

typedef __attribute__((ext_vector_type(16))) _Float16 v16h;
typedef __attribute__((ext_vector_type(8)))  float    v8f;

#define Bsz   2048
#define Hh    512
#define Ss    128
#define Tt    8
#define Ff    8
#define FS    (Ff * Ss)   // 1024
#define KSTEPS (Hh / 32)  // 16

#if defined(__HIP_DEVICE_COMPILE__) && \
    __has_builtin(__builtin_amdgcn_global_load_async_to_lds_b128) && \
    __has_builtin(__builtin_amdgcn_s_wait_asynccnt)
#define USE_ASYNC_LDS 1
#else
#define USE_ASYNC_LDS 0
#endif

#if USE_ASYNC_LDS
// Builtin signature (probe-confirmed): (int4 addrspace(1)*, int4 addrspace(3)*,
// imm offset, imm cpol). Clang prints AS1 as "__device__", AS3 as "__shared__".
typedef int i32x4_t __attribute__((vector_size(16)));
typedef __attribute__((address_space(1))) i32x4_t g1_i32x4;
typedef __attribute__((address_space(3))) i32x4_t l3_i32x4;
#endif

union Frag {
  v16h v;
  uint4 u[2];
};

// A-matrix fragment (16x32 f16, row-major, ld in halves).
// Lane L: row m = L&15, koff = (L>=16)?8:0; halves 0..7 -> k[koff..koff+8),
// halves 8..15 -> k[16+koff..16+koff+8).
__device__ __forceinline__ Frag load_a(const _Float16* base, int ld, int lane) {
  const _Float16* p = base + (size_t)(lane & 15) * ld + ((lane >> 4) << 3);
  Frag f;
  f.u[0] = *(const uint4*)(p);
  f.u[1] = *(const uint4*)(p + 16);
  return f;
}

// Same A layout but from an LDS tile [16][32] (ds_load_b128 x2).
__device__ __forceinline__ Frag load_a_lds(const _Float16* base, int lane) {
  const _Float16* p = base + ((lane & 15) << 5) + ((lane >> 4) << 3);
  Frag f;
  f.u[0] = *(const uint4*)(p);
  f.u[1] = *(const uint4*)(p + 16);
  return f;
}

// B-matrix fragment (32x16 f16) from k-major (pre-transposed) weights
// Wt[k*ld + n]. Lane L holds column n = L&15, koff = (L>=16)?8:0; halves 0..7 ->
// k[koff..koff+8), halves 8..15 -> k[16+koff..16+koff+8) (same as A).
__device__ __forceinline__ Frag load_b(const _Float16* base, int ld, int lane) {
  const _Float16* p = base + (lane & 15) + (size_t)((lane >> 4) << 3) * ld;
  Frag f;
#pragma unroll
  for (int s = 0; s < 8; ++s) {
    f.v[s] = p[(size_t)s * ld];
    f.v[8 + s] = p[(size_t)(16 + s) * ld];
  }
  return f;
}

// 16-byte global -> LDS copy: async (ASYNCcnt) on gfx1250, sync fallback.
__device__ __forceinline__ void copy16_g2l(const _Float16* g, _Float16* l) {
#if USE_ASYNC_LDS
  __builtin_amdgcn_global_load_async_to_lds_b128((g1_i32x4*)g, (l3_i32x4*)l, 0, 0);
#else
  *(uint4*)l = *(const uint4*)g;
#endif
}

__device__ __forceinline__ void wait_async_le2() {
#if USE_ASYNC_LDS
  __builtin_amdgcn_s_wait_asynccnt(2);
#endif
}
__device__ __forceinline__ void wait_async_le4() {
#if USE_ASYNC_LDS
  __builtin_amdgcn_s_wait_asynccnt(4);
#endif
}
__device__ __forceinline__ void wait_async_0() {
#if USE_ASYNC_LDS
  __builtin_amdgcn_s_wait_asynccnt(0);
#endif
}

// One 16B chunk of a (T=8 x 16 x 32) A-tile stage; c in [0, 512).
__device__ __forceinline__ void stage_chunk(const _Float16* __restrict__ g,
                                            _Float16* lds, int c) {
  const int t  = c >> 6;
  const int m  = (c >> 2) & 15;
  const int kc = (c & 3) << 3;
  copy16_g2l(g + ((size_t)t * Bsz + m) * Hh + kc,
             lds + ((t << 9) + (m << 5) + kc));
}

// Stage one K-step of spike A-tiles for all T=8 timesteps into LDS:
// 8KB = 512 chunks of 16B; 256 threads x exactly 2 chunks, unconditional
// (=> 2 async ops per wave, no EXEC-mask branches around the issues).
// g points at (row b0, col k0): element (t,m,kc) at g + ((t*Bsz+m)*Hh + kc).
__device__ __forceinline__ void stage_A(const _Float16* __restrict__ g,
                                        _Float16* lds /* [8][16][32] */,
                                        int tid) {
  stage_chunk(g, lds, tid);
  stage_chunk(g, lds, tid + 256);
}

// ---------------------------------------------------------------------------
// Prep kernels
// ---------------------------------------------------------------------------
__global__ void k_f32_to_f16(const float* __restrict__ in,
                             _Float16* __restrict__ out, long n) {
  long i = (long)blockIdx.x * blockDim.x + threadIdx.x;
  if (i < n) out[i] = (_Float16)in[i];
}

// Wt[i*outF + o] = (f16) W[o*inF + i]   (row-major (out,in) -> k-major)
__global__ void k_transpose_to_f16(const float* __restrict__ W,
                                   _Float16* __restrict__ Wt,
                                   int outF, int inF) {
  long idx = (long)blockIdx.x * blockDim.x + threadIdx.x;
  if (idx < (long)outF * inF) {
    int o = (int)(idx / inF);
    int i = (int)(idx % inF);
    Wt[(size_t)i * outF + o] = (_Float16)W[idx];
  }
}

__global__ void k_zero_u4(uint4* __restrict__ p, long n) {
  long i = (long)blockIdx.x * blockDim.x + threadIdx.x;
  long stride = (long)gridDim.x * blockDim.x;
  uint4 z = make_uint4(0u, 0u, 0u, 0u);
  for (; i < n; i += stride) p[i] = z;
}

// ---------------------------------------------------------------------------
// RNN1 one frame, fused with the i2h GEMM:
//   ih   = state[:,f,:] @ W_i1^T + b_i1 + b_h1        (t-invariant, K=128)
//   x(t) = ih + h1_prev(t) @ W_h1^T                   (K=512, LDS-staged A)
//   LIF over T (tau=2, v_th=1, hard reset) -> h1_cur spikes (f16 {0,1}).
// grid (B/16, H/128), block 256 = 8 waves; wave -> one 16x16 (b,n) tile
// with 8 per-timestep accumulators sharing each B fragment.
// ---------------------------------------------------------------------------
__global__ void __launch_bounds__(256)
k_rnn1(const _Float16* __restrict__ state_h,  // (B, F*S)
       const _Float16* __restrict__ Wi1_t,    // (S, H) k-major
       const _Float16* __restrict__ h1_prev,  // (T,B,H) spikes
       const _Float16* __restrict__ Wh1_t,    // (H, H) k-major
       const float* __restrict__ b_i1,
       const float* __restrict__ b_h1,
       _Float16* __restrict__ h1_cur,         // (T,B,H) spikes
       int frame) {
  __shared__ __align__(16) _Float16 As[2][Tt][16][32];   // 16 KB

  const int tid  = threadIdx.x;
  const int lane = tid & 31;
  const int wv   = tid >> 5;
  const int b0   = blockIdx.x << 4;
  const int n0   = (blockIdx.y << 7) + (wv << 4);
  const int n    = lane & 15;
  const int m0   = (lane >> 4) << 3;

  // ---- i2h part: one accumulator, A shared across t ----
  v8f ih;
  const float bias = b_i1[n0 + n] + b_h1[n0 + n];
#pragma unroll
  for (int r = 0; r < 8; ++r) ih[r] = bias;

  const _Float16* Astate = state_h + (size_t)b0 * FS + (size_t)frame * Ss;
#pragma unroll
  for (int k0 = 0; k0 < Ss; k0 += 32) {
    Frag a = load_a(Astate + k0, FS, lane);
    Frag b = load_b(Wi1_t + (size_t)k0 * Hh + n0, Hh, lane);
    ih = __builtin_amdgcn_wmma_f32_16x16x32_f16(false, a.v, false, b.v,
                                                (short)0, ih, false, false);
  }

  v8f acc[Tt];
#pragma unroll
  for (int t = 0; t < Tt; ++t) acc[t] = ih;

  // ---- recurrent part: K=512, A staged in LDS (double-buffered async) ----
  const _Float16* Gp = h1_prev + (size_t)b0 * Hh;
  stage_A(Gp, &As[0][0][0][0], tid);
  for (int ks = 0; ks < KSTEPS; ++ks) {
    const int cur = ks & 1;
    if (ks + 1 < KSTEPS) {
      stage_A(Gp + (ks + 1) * 32, &As[cur ^ 1][0][0][0], tid);
      wait_async_le2();            // cur's 2 copies done; next 2 in flight
    } else {
      wait_async_0();
    }
    __syncthreads();

    Frag bf = load_b(Wh1_t + (size_t)(ks * 32) * Hh + n0, Hh, lane);
    if (ks + 1 < KSTEPS)
      __builtin_prefetch(Wh1_t + (size_t)((ks + 1) * 32) * Hh + n0, 0, 0);
#pragma unroll
    for (int t = 0; t < Tt; ++t) {
      Frag af = load_a_lds(&As[cur][t][0][0], lane);
      acc[t] = __builtin_amdgcn_wmma_f32_16x16x32_f16(false, af.v, false, bf.v,
                                                      (short)0, acc[t], false, false);
    }
    __syncthreads();               // protect buffer before next overwrite
  }

  // ---- LIF scan over T, emit spikes ----
#pragma unroll
  for (int r = 0; r < 8; ++r) {
    float v = 0.f;
#pragma unroll
    for (int t = 0; t < Tt; ++t) {
      v += (acc[t][r] - v) * 0.5f;
      float s = (v >= 1.0f) ? 1.f : 0.f;
      h1_cur[((size_t)t * Bsz + b0 + m0 + r) * Hh + n0 + n] = (_Float16)s;
      v *= (1.f - s);
    }
  }
}

// ---------------------------------------------------------------------------
// RNN2 one frame: x(t) = h1(t) @ W_i2^T + b_i2 + h2_prev(t) @ W_h2^T + b_h2,
// LIF(v_th=999); writes post-reset final membrane potential (B,H) -> v_last.
// Both A-operands staged in LDS (2 x 8KB x double buffer = 32KB).
// ---------------------------------------------------------------------------
__global__ void __launch_bounds__(256)
k_rnn2(const _Float16* __restrict__ h1_cur,   // (T,B,H)
       const _Float16* __restrict__ Wi2_t,    // (H,H) k-major
       const _Float16* __restrict__ h2_prev,  // (T,B,H)
       const _Float16* __restrict__ Wh2_t,    // (H,H) k-major
       const float* __restrict__ b_i2,
       const float* __restrict__ b_h2,
       _Float16* __restrict__ h2_cur,         // (T,B,H)
       float* __restrict__ v_last) {          // (B,H)
  __shared__ __align__(16) _Float16 As1[2][Tt][16][32];  // 16 KB
  __shared__ __align__(16) _Float16 As2[2][Tt][16][32];  // 16 KB

  const int tid  = threadIdx.x;
  const int lane = tid & 31;
  const int wv   = tid >> 5;
  const int b0   = blockIdx.x << 4;
  const int n0   = (blockIdx.y << 7) + (wv << 4);
  const int n    = lane & 15;
  const int m0   = (lane >> 4) << 3;

  const float bias = b_i2[n0 + n] + b_h2[n0 + n];
  v8f acc[Tt];
#pragma unroll
  for (int t = 0; t < Tt; ++t)
#pragma unroll
    for (int r = 0; r < 8; ++r) acc[t][r] = bias;

  const _Float16* G1 = h1_cur + (size_t)b0 * Hh;
  const _Float16* G2 = h2_prev + (size_t)b0 * Hh;
  stage_A(G1, &As1[0][0][0][0], tid);
  stage_A(G2, &As2[0][0][0][0], tid);
  for (int ks = 0; ks < KSTEPS; ++ks) {
    const int cur = ks & 1;
    if (ks + 1 < KSTEPS) {
      stage_A(G1 + (ks + 1) * 32, &As1[cur ^ 1][0][0][0], tid);
      stage_A(G2 + (ks + 1) * 32, &As2[cur ^ 1][0][0][0], tid);
      wait_async_le4();            // cur's 4 copies done; next 4 in flight
    } else {
      wait_async_0();
    }
    __syncthreads();

    Frag bi = load_b(Wi2_t + (size_t)(ks * 32) * Hh + n0, Hh, lane);
    Frag bh = load_b(Wh2_t + (size_t)(ks * 32) * Hh + n0, Hh, lane);
    if (ks + 1 < KSTEPS) {
      __builtin_prefetch(Wi2_t + (size_t)((ks + 1) * 32) * Hh + n0, 0, 0);
      __builtin_prefetch(Wh2_t + (size_t)((ks + 1) * 32) * Hh + n0, 0, 0);
    }
#pragma unroll
    for (int t = 0; t < Tt; ++t) {
      Frag a1 = load_a_lds(&As1[cur][t][0][0], lane);
      acc[t] = __builtin_amdgcn_wmma_f32_16x16x32_f16(false, a1.v, false, bi.v,
                                                      (short)0, acc[t], false, false);
      Frag a2 = load_a_lds(&As2[cur][t][0][0], lane);
      acc[t] = __builtin_amdgcn_wmma_f32_16x16x32_f16(false, a2.v, false, bh.v,
                                                      (short)0, acc[t], false, false);
    }
    __syncthreads();
  }

#pragma unroll
  for (int r = 0; r < 8; ++r) {
    float v = 0.f;
#pragma unroll
    for (int t = 0; t < Tt; ++t) {
      v += (acc[t][r] - v) * 0.5f;
      float s = (v >= 999.0f) ? 1.f : 0.f;
      h2_cur[((size_t)t * Bsz + b0 + m0 + r) * Hh + n0 + n] = (_Float16)s;
      v *= (1.f - s);
    }
    v_last[(size_t)(b0 + m0 + r) * Hh + n0 + n] = v;  // last frame's write wins
  }
}

// ---------------------------------------------------------------------------
// MLP head: q[b] = relu([v_last[b], action[b]] @ W_f1^T + b_f1) @ W_f2^T + b_f2
// ---------------------------------------------------------------------------
__global__ void k_final_mlp(const float* __restrict__ v_last,  // (B,H)
                            const float* __restrict__ action,  // (B,4)
                            const float* __restrict__ W_f1,    // (H, H+4)
                            const float* __restrict__ b_f1,    // (H)
                            const float* __restrict__ W_f2,    // (1, H)
                            const float* __restrict__ b_f2,    // (1)
                            float* __restrict__ q) {           // (B)
  __shared__ float xs[Hh + 4];
  __shared__ float partial[256];
  const int b   = blockIdx.x;
  const int tid = threadIdx.x;

  for (int i = tid; i < Hh; i += 256) xs[i] = v_last[(size_t)b * Hh + i];
  if (tid < 4) xs[Hh + tid] = action[(size_t)b * 4 + tid];
  __syncthreads();

  float local = 0.f;
  for (int j = tid; j < Hh; j += 256) {
    const float* w = W_f1 + (size_t)j * (Hh + 4);
    float acc = b_f1[j];
#pragma unroll 4
    for (int i = 0; i < Hh + 4; ++i) acc = fmaf(w[i], xs[i], acc);
    acc = fmaxf(acc, 0.f);
    local = fmaf(acc, W_f2[j], local);
  }
  partial[tid] = local;
  __syncthreads();
  for (int s = 128; s > 0; s >>= 1) {
    if (tid < s) partial[tid] += partial[tid + s];
    __syncthreads();
  }
  if (tid == 0) q[b] = partial[0] + b_f2[0];
}

// ---------------------------------------------------------------------------
// Host launcher
// ---------------------------------------------------------------------------
extern "C" void kernel_launch(void* const* d_in, const int* in_sizes, int n_in,
                              void* d_out, int out_size, void* d_ws, size_t ws_size,
                              hipStream_t stream) {
  (void)in_sizes; (void)n_in; (void)out_size; (void)ws_size;

  const float* state  = (const float*)d_in[0];   // (B,F,S)
  const float* action = (const float*)d_in[1];   // (B,4)
  const float* W_i1   = (const float*)d_in[2];   // (H,S)
  const float* b_i1   = (const float*)d_in[3];
  const float* W_h1   = (const float*)d_in[4];   // (H,H)
  const float* b_h1   = (const float*)d_in[5];
  const float* W_i2   = (const float*)d_in[6];   // (H,H)
  const float* b_i2   = (const float*)d_in[7];
  const float* W_h2   = (const float*)d_in[8];   // (H,H)
  const float* b_h2   = (const float*)d_in[9];
  const float* W_f1   = (const float*)d_in[10];  // (H,H+4)
  const float* b_f1   = (const float*)d_in[11];
  const float* W_f2   = (const float*)d_in[12];  // (1,H)
  const float* b_f2   = (const float*)d_in[13];
  float* q_out = (float*)d_out;                  // (B,1)

  // --- workspace carve (all sizes 256B multiples) ---
  char* p = (char*)d_ws;
  auto carve = [&](size_t bytes) -> void* {
    void* r = (void*)p;
    p += (bytes + 255) & ~(size_t)255;
    return r;
  };
  _Float16* state_h = (_Float16*)carve((size_t)Bsz * FS * 2);   // 4 MB
  _Float16* Wi1_t   = (_Float16*)carve((size_t)Ss * Hh * 2);    // 128 KB
  _Float16* Wh1_t   = (_Float16*)carve((size_t)Hh * Hh * 2);    // 512 KB
  _Float16* Wi2_t   = (_Float16*)carve((size_t)Hh * Hh * 2);
  _Float16* Wh2_t   = (_Float16*)carve((size_t)Hh * Hh * 2);
  const size_t spikeElems = (size_t)Tt * Bsz * Hh;              // 8M
  _Float16* h1a = (_Float16*)carve(spikeElems * 2);             // 16 MB
  _Float16* h2a = (_Float16*)carve(spikeElems * 2);             // contig w/ h1a
  _Float16* h1b = (_Float16*)carve(spikeElems * 2);
  _Float16* h2b = (_Float16*)carve(spikeElems * 2);
  float* vlast  = (float*)carve((size_t)Bsz * Hh * 4);          // 4 MB

  // --- prep (re-run every call: deterministic) ---
  {
    long n = (long)Bsz * FS;
    k_f32_to_f16<<<(unsigned)(n / 256), 256, 0, stream>>>(state, state_h, n);
  }
  k_transpose_to_f16<<<(Hh * Ss) / 256, 256, 0, stream>>>(W_i1, Wi1_t, Hh, Ss);
  k_transpose_to_f16<<<(Hh * Hh) / 256, 256, 0, stream>>>(W_h1, Wh1_t, Hh, Hh);
  k_transpose_to_f16<<<(Hh * Hh) / 256, 256, 0, stream>>>(W_i2, Wi2_t, Hh, Hh);
  k_transpose_to_f16<<<(Hh * Hh) / 256, 256, 0, stream>>>(W_h2, Wh2_t, Hh, Hh);
  {
    // zero initial spike carries (h1a + h2a contiguous: 32 MB)
    long nu4 = (long)(2 * spikeElems * 2) / 16;
    k_zero_u4<<<8192, 256, 0, stream>>>((uint4*)h1a, nu4);
  }

  // --- frames (serial dependency), ping-pong spike buffers ---
  dim3 grid(Bsz / 16, Hh / 128);  // 128 x 4 blocks, 8 waves each
  for (int f = 0; f < Ff; ++f) {
    _Float16* h1_prev = (f & 1) ? h1b : h1a;
    _Float16* h1_cur  = (f & 1) ? h1a : h1b;
    _Float16* h2_prev = (f & 1) ? h2b : h2a;
    _Float16* h2_cur  = (f & 1) ? h2a : h2b;

    k_rnn1<<<grid, 256, 0, stream>>>(state_h, Wi1_t, h1_prev, Wh1_t,
                                     b_i1, b_h1, h1_cur, f);
    k_rnn2<<<grid, 256, 0, stream>>>(h1_cur, Wi2_t, h2_prev, Wh2_t,
                                     b_i2, b_h2, h2_cur, vlast);
  }

  // --- MLP head ---
  k_final_mlp<<<Bsz, 256, 0, stream>>>(vlast, action, W_f1, b_f1, W_f2, b_f2, q_out);
}
